// simpleGNN_73091753443468
// MI455X (gfx1250) — matrix operand order repaired and checked by reading, not compiled, not run on hardware
//
#include <hip/hip_runtime.h>

typedef _Float16 h8   __attribute__((ext_vector_type(8)));
typedef _Float16 v16h __attribute__((ext_vector_type(16)));
typedef float    v8f  __attribute__((ext_vector_type(8)));

#define GH   3          // attention heads
#define GHD  512        // hidden dim
#define GDIN 256        // input dim
#define NEG_SLOPE 0.2f
#define BN_EPS 1e-5f

// ---------------- tiled WMMA GEMM: C[M,Nout] = A[M,K] (f16) * W[Nout,K]^T (f16), f32 accum ----------------
// Double-buffered LDS tiles filled with GLOBAL_LOAD_ASYNC_TO_LDS_B128 (ASYNCcnt),
// one s_wait_asynccnt + one barrier per K-step; WMMA consumes tile t while tile t+1 streams in.
#define TBM 64
#define TBN 128
#define TBK 32
#define AST 40   // LDS row stride (elements), 80B = 16B aligned, conflict-padded
#define BST 40

__global__ __launch_bounds__(256) void gnn_gemm_wmma_f16(
    const _Float16* __restrict__ A, const _Float16* __restrict__ W,
    const _Float16* __restrict__ zpad,   // >=16B of zeros for M-tail rows
    float* __restrict__ C, int M, int K, int Nout)
{
  __shared__ _Float16 As[2][TBM * AST];
  __shared__ _Float16 Bs[2][TBN * BST];

  const int tid  = threadIdx.x;
  const int lane = tid & 31;
  const int wave = tid >> 5;
  const int hl   = lane >> 4;    // lane half (0: lanes 0-15, 1: lanes 16-31)
  const int l16  = lane & 15;
  const int wm   = wave & 3;     // wave m-tile (0..3)
  const int wn   = wave >> 2;    // wave n-strip (0..1)

  const int m0 = blockIdx.x * TBM;
  const int n0 = blockIdx.y * TBN;

  v8f acc[4] = {};

  // tile-load coordinates: 4 threads per row, one 16B async transfer per row-quarter
  const int tr = tid >> 2;        // 0..63
  const int tc = (tid & 3) << 3;  // 0,8,16,24
  const int gm = m0 + tr;
  const _Float16* arow = (gm < M) ? (A + (size_t)gm * K) : nullptr;
  const _Float16* brow0 = W + (size_t)(n0 + tr) * K;        // Nout multiple of 128: no guard
  const _Float16* brow1 = W + (size_t)(n0 + tr + 64) * K;

  // async fill of one K-tile into LDS buffer `buf`
  auto issue_tile = [&](int k0, int buf) {
    const _Float16* ga = arow ? (arow + k0 + tc) : zpad;    // tail rows pull zeros
    unsigned la = (unsigned)(size_t)&As[buf][tr * AST + tc];
    asm volatile("global_load_async_to_lds_b128 %0, %1, off" :: "v"(la), "v"(ga) : "memory");
    unsigned lb0 = (unsigned)(size_t)&Bs[buf][tr * BST + tc];
    unsigned lb1 = (unsigned)(size_t)&Bs[buf][(tr + 64) * BST + tc];
    const _Float16* gb0 = brow0 + k0 + tc;
    const _Float16* gb1 = brow1 + k0 + tc;
    asm volatile("global_load_async_to_lds_b128 %0, %1, off" :: "v"(lb0), "v"(gb0) : "memory");
    asm volatile("global_load_async_to_lds_b128 %0, %1, off" :: "v"(lb1), "v"(gb1) : "memory");
    if (arow && k0 + TBK < K) __builtin_prefetch(arow + k0 + TBK + tc, 0, 0);
  };

  const int nt = K / TBK;
  issue_tile(0, 0);

  for (int t = 0; t < nt; ++t) {
    const int buf = t & 1;
    asm volatile("s_wait_asynccnt 0x0" ::: "memory");  // my tile-t transfers landed in LDS
    __syncthreads();                                    // everyone's landed; prev compute done
    if (t + 1 < nt) issue_tile((t + 1) * TBK, 1 - buf);

    // A fragment (ISA 7.12.2): half0 lanes carry K 0-7 / 16-23, half1 lanes K 8-15 / 24-31
    const _Float16* ap = &As[buf][(wm * 16 + l16) * AST];
    h8 alo = *reinterpret_cast<const h8*>(ap + hl * 8);
    h8 ahi = *reinterpret_cast<const h8*>(ap + 16 + hl * 8);
    v16h afrag = __builtin_shufflevector(alo, ahi, 0,1,2,3,4,5,6,7,8,9,10,11,12,13,14,15);

    // preload all 4 B fragments (lane = column n; half0 lanes K 0-7 / 16-23, half1 lanes K 8-15 / 24-31)
    v16h bfr[4];
    #pragma unroll
    for (int s = 0; s < 4; ++s) {
      const _Float16* bp = &Bs[buf][(wn * 64 + s * 16 + l16) * BST];
      h8 blo = *reinterpret_cast<const h8*>(bp + hl * 8);
      h8 bhi = *reinterpret_cast<const h8*>(bp + 16 + hl * 8);
      bfr[s] = __builtin_shufflevector(blo, bhi, 0,1,2,3,4,5,6,7,8,9,10,11,12,13,14,15);
    }
    #pragma unroll
    for (int s = 0; s < 4; ++s)
      acc[s] = __builtin_amdgcn_wmma_f32_16x16x32_f16(
          false, afrag, false, bfr[s], (short)0, acc[s], false, false);
  }

  // C layout: lane half selects M base (+8), vgpr index i is M offset, l16 is N
  #pragma unroll
  for (int s = 0; s < 4; ++s) {
    const int n = n0 + wn * 64 + s * 16 + l16;
    #pragma unroll
    for (int i = 0; i < 8; ++i) {
      const int m = m0 + wm * 16 + hl * 8 + i;
      if (m < M) C[(size_t)m * Nout + n] = acc[s][i];
    }
  }
}

// ---------------- elementwise / attention helpers ----------------

__global__ void gnn_zero_f32(float* __restrict__ p, int n) {
  int i = blockIdx.x * blockDim.x + threadIdx.x;
  if (i < n) p[i] = 0.f;
}

__global__ void gnn_cast_f16(const float* __restrict__ in, _Float16* __restrict__ out, int n) {
  int i = blockIdx.x * blockDim.x + threadIdx.x;
  if (i < n) out[i] = (_Float16)in[i];
}

__device__ __forceinline__ unsigned f2sortable(float f) {
  unsigned u = __float_as_uint(f);
  return (u & 0x80000000u) ? ~u : (u | 0x80000000u);
}
__device__ __forceinline__ float sortable2f(unsigned s) {
  unsigned u = (s & 0x80000000u) ? (s & 0x7FFFFFFFu) : ~s;
  return __uint_as_float(u);
}

// per-node attention scores: a_s[n,h] = <h[n,h,:], att_src[h,:]>, same for dst
__global__ __launch_bounds__(128) void gnn_attn_scores(
    const float* __restrict__ hbuf, const float* __restrict__ att_s,
    const float* __restrict__ att_d, float* __restrict__ a_s, float* __restrict__ a_d)
{
  __shared__ float red[128];
  const int n = blockIdx.x, tid = threadIdx.x;
  for (int hh = 0; hh < GH; ++hh) {
    float ss = 0.f, dd = 0.f;
    const float* hp = hbuf + (size_t)n * (GH * GHD) + hh * GHD;
    for (int d = tid; d < GHD; d += 128) {
      float v = hp[d];
      ss += v * att_s[hh * GHD + d];
      dd += v * att_d[hh * GHD + d];
    }
    red[tid] = ss; __syncthreads();
    for (int o = 64; o > 0; o >>= 1) { if (tid < o) red[tid] += red[tid + o]; __syncthreads(); }
    if (tid == 0) a_s[n * GH + hh] = red[0];
    __syncthreads();
    red[tid] = dd; __syncthreads();
    for (int o = 64; o > 0; o >>= 1) { if (tid < o) red[tid] += red[tid + o]; __syncthreads(); }
    if (tid == 0) a_d[n * GH + hh] = red[0];
    __syncthreads();
  }
}

__global__ void gnn_init_attn(unsigned* __restrict__ mmax, float* __restrict__ den, int n3) {
  int i = blockIdx.x * blockDim.x + threadIdx.x;
  if (i < n3) { mmax[i] = 0u; den[i] = 0.f; }  // 0 < f2sortable(-inf): safe -inf floor
}

__global__ void gnn_init_bias(float* __restrict__ agg, const float* __restrict__ bias, int total, int C) {
  int i = blockIdx.x * blockDim.x + threadIdx.x;
  if (i < total) agg[i] = bias[i % C];
}

__device__ __forceinline__ void edge_nodes(const long long* ei, int E, int e, int& s, int& d) {
  if (e < E) { s = (int)ei[e]; d = (int)ei[E + e]; } else { s = d = e - E; }
}

__global__ void gnn_edge_max(const long long* __restrict__ ei, int E, int TE,
                             const float* __restrict__ a_s, const float* __restrict__ a_d,
                             unsigned* __restrict__ mmax)
{
  int e = blockIdx.x * blockDim.x + threadIdx.x;
  if (e >= TE) return;
  int s, d; edge_nodes(ei, E, e, s, d);
  #pragma unroll
  for (int hh = 0; hh < GH; ++hh) {
    float v = a_s[s * GH + hh] + a_d[d * GH + hh];
    v = v > 0.f ? v : NEG_SLOPE * v;
    atomicMax(&mmax[d * GH + hh], f2sortable(v));
  }
}

__global__ void gnn_edge_exp(const long long* __restrict__ ei, int E, int TE,
                             const float* __restrict__ a_s, const float* __restrict__ a_d,
                             const unsigned* __restrict__ mmax, float* __restrict__ den,
                             float* __restrict__ ex)
{
  int e = blockIdx.x * blockDim.x + threadIdx.x;
  if (e >= TE) return;
  int s, d; edge_nodes(ei, E, e, s, d);
  #pragma unroll
  for (int hh = 0; hh < GH; ++hh) {
    float v = a_s[s * GH + hh] + a_d[d * GH + hh];
    v = v > 0.f ? v : NEG_SLOPE * v;
    float m = sortable2f(mmax[d * GH + hh]);
    float x = __expf(v - m);
    ex[e * GH + hh] = x;
    atomicAdd(&den[d * GH + hh], x);
  }
}

// one block per edge: agg[dst,:] += sum_h alpha[e,h] * h[src,h,:] / GH
__global__ __launch_bounds__(256) void gnn_edge_aggregate(
    const long long* __restrict__ ei, int E,
    const float* __restrict__ ex, const float* __restrict__ den,
    const float* __restrict__ hbuf, float* __restrict__ agg)
{
  int e = blockIdx.x;
  int s, d; edge_nodes(ei, E, e, s, d);
  float al0 = ex[e * GH + 0] / den[d * GH + 0];
  float al1 = ex[e * GH + 1] / den[d * GH + 1];
  float al2 = ex[e * GH + 2] / den[d * GH + 2];
  const float* hp = hbuf + (size_t)s * (GH * GHD);
  float* op = agg + (size_t)d * GHD;
  for (int c = threadIdx.x; c < GHD; c += 256) {
    float v = al0 * hp[c] + al1 * hp[GHD + c] + al2 * hp[2 * GHD + c];
    atomicAdd(&op[c], v * (1.f / (float)GH));
  }
}

// batch-norm column statistics (population var, matching jnp.var)
__global__ __launch_bounds__(256) void gnn_col_stats(
    const float* __restrict__ X, float* __restrict__ mean, float* __restrict__ rstd, int N, int C)
{
  __shared__ float rs[256], rq[256];
  const int c = blockIdx.x, tid = threadIdx.x;
  float s = 0.f, q = 0.f;
  for (int r = tid; r < N; r += 256) {
    float v = X[(size_t)r * C + c];
    s += v; q += v * v;
  }
  rs[tid] = s; rq[tid] = q; __syncthreads();
  for (int o = 128; o > 0; o >>= 1) {
    if (tid < o) { rs[tid] += rs[tid + o]; rq[tid] += rq[tid + o]; }
    __syncthreads();
  }
  if (tid == 0) {
    float mu = rs[0] / (float)N;
    float var = rq[0] / (float)N - mu * mu;
    mean[c] = mu;
    rstd[c] = rsqrtf(var + BN_EPS);
  }
}

__global__ void gnn_bn_relu_cast(const float* __restrict__ X,
                                 const float* __restrict__ mean, const float* __restrict__ rstd,
                                 const float* __restrict__ gamma, const float* __restrict__ beta,
                                 _Float16* __restrict__ out, int total, int C)
{
  int i = blockIdx.x * blockDim.x + threadIdx.x;
  if (i >= total) return;
  int c = i % C;
  float v = (X[i] - mean[c]) * rstd[c] * gamma[c] + beta[c];
  out[i] = (_Float16)(v > 0.f ? v : 0.f);
}

__global__ void gnn_final(const float* __restrict__ X, const float* __restrict__ db2,
                          const float* __restrict__ sf, float* __restrict__ out, int N)
{
  int i = blockIdx.x * blockDim.x + threadIdx.x;
  int total = N * GDIN;
  if (i >= total) return;
  int c = i % GDIN, n = i / GDIN;
  float v = X[i] + db2[c];
  v = v > 0.f ? v : 0.f;
  out[i] = v * sf[n];
}

// ---------------- host launcher ----------------
extern "C" void kernel_launch(void* const* d_in, const int* in_sizes, int n_in,
                              void* d_out, int out_size, void* d_ws, size_t ws_size,
                              hipStream_t stream)
{
  (void)n_in; (void)out_size; (void)ws_size;
  const float*     x   = (const float*)d_in[0];
  const long long* ei  = (const long long*)d_in[1];
  const float*     sf  = (const float*)d_in[2];
  const float*     W1  = (const float*)d_in[3];
  const float*     as1 = (const float*)d_in[4];
  const float*     ad1 = (const float*)d_in[5];
  const float*     b1  = (const float*)d_in[6];
  const float*     g1  = (const float*)d_in[7];
  const float*     be1 = (const float*)d_in[8];
  const float*     W2  = (const float*)d_in[9];
  const float*     as2 = (const float*)d_in[10];
  const float*     ad2 = (const float*)d_in[11];
  const float*     b2  = (const float*)d_in[12];
  const float*     g2  = (const float*)d_in[13];
  const float*     be2 = (const float*)d_in[14];
  const float*     dw1 = (const float*)d_in[15];
  // d_in[16] (db1) feeds a BatchNorm input: it shifts mean identically and cancels exactly.
  const float*     dg1 = (const float*)d_in[17];
  const float*     dbe1= (const float*)d_in[18];
  const float*     dw2 = (const float*)d_in[19];
  const float*     db2 = (const float*)d_in[20];

  const int N  = in_sizes[0] / GDIN;
  const int E  = in_sizes[1] / 2;
  const int TE = E + N;          // edges + self loops
  const int HN = GH * GHD;       // 1536

  // workspace layout (256B aligned)
  char* base = (char*)d_ws;
  auto au = [](size_t v) { return (v + 255) & ~(size_t)255; };
  size_t off = 0;
  _Float16* xb   = (_Float16*)(base + off); off += au((size_t)N * GHD * 2);      // f16 activations (max K=512)
  _Float16* wb   = (_Float16*)(base + off); off += au((size_t)HN * GHD * 2);     // f16 weights (max 1536x512)
  float*    hbuf = (float*)(base + off);    off += au((size_t)N * HN * 4);       // GEMM output [N,1536]
  float*    agg  = (float*)(base + off);    off += au((size_t)N * GHD * 4);      // aggregated GAT output
  float*    a_s  = (float*)(base + off);    off += au((size_t)N * GH * 4);
  float*    a_d  = (float*)(base + off);    off += au((size_t)N * GH * 4);
  unsigned* mmax = (unsigned*)(base + off); off += au((size_t)N * GH * 4);
  float*    den  = (float*)(base + off);    off += au((size_t)N * GH * 4);
  float*    exb  = (float*)(base + off);    off += au((size_t)TE * GH * 4);
  float*    mean = (float*)(base + off);    off += au((size_t)GHD * 4);
  float*    rstd = (float*)(base + off);    off += au((size_t)GHD * 4);
  float*    zpad = (float*)(base + off);    off += au((size_t)64 * 4);           // zero region for GEMM tail rows

  auto cdiv = [](int a, int b) { return (a + b - 1) / b; };

  gnn_zero_f32<<<1, 64, 0, stream>>>(zpad, 64);   // async-load target for out-of-range A rows

  auto gemm = [&](const _Float16* Aact, const _Float16* Wf16, float* Cout, int K, int Nout) {
    dim3 g(cdiv(N, TBM), Nout / TBN);
    gnn_gemm_wmma_f16<<<g, 256, 0, stream>>>(Aact, Wf16, (const _Float16*)zpad, Cout, N, K, Nout);
  };

  auto gat_layer = [&](const _Float16* act, int K, const float* Wsrc,
                       const float* att_s, const float* att_d, const float* bias) {
    gnn_cast_f16<<<cdiv(HN * K, 256), 256, 0, stream>>>(Wsrc, wb, HN * K);
    gemm(act, wb, hbuf, K, HN);
    gnn_attn_scores<<<N, 128, 0, stream>>>(hbuf, att_s, att_d, a_s, a_d);
    gnn_init_attn<<<cdiv(N * GH, 256), 256, 0, stream>>>(mmax, den, N * GH);
    gnn_init_bias<<<cdiv(N * GHD, 256), 256, 0, stream>>>(agg, bias, N * GHD, GHD);
    gnn_edge_max<<<cdiv(TE, 256), 256, 0, stream>>>(ei, E, TE, a_s, a_d, mmax);
    gnn_edge_exp<<<cdiv(TE, 256), 256, 0, stream>>>(ei, E, TE, a_s, a_d, mmax, den, exb);
    gnn_edge_aggregate<<<TE, 256, 0, stream>>>(ei, E, exb, den, hbuf, agg);
  };

  // ---- GAT layer 1 (x: f32 [N,256] -> f16) ----
  gnn_cast_f16<<<cdiv(N * GDIN, 256), 256, 0, stream>>>(x, xb, N * GDIN);
  gat_layer(xb, GDIN, W1, as1, ad1, b1);
  gnn_col_stats<<<GHD, 256, 0, stream>>>(agg, mean, rstd, N, GHD);
  gnn_bn_relu_cast<<<cdiv(N * GHD, 256), 256, 0, stream>>>(agg, mean, rstd, g1, be1, xb, N * GHD, GHD);

  // ---- GAT layer 2 ----
  gat_layer(xb, GHD, W2, as2, ad2, b2);
  gnn_col_stats<<<GHD, 256, 0, stream>>>(agg, mean, rstd, N, GHD);
  gnn_bn_relu_cast<<<cdiv(N * GHD, 256), 256, 0, stream>>>(agg, mean, rstd, g2, be2, xb, N * GHD, GHD);

  // ---- dense head 1: z @ dw1^T (+db1 cancels in BN), BN, ReLU ----
  gnn_cast_f16<<<cdiv(GHD * GHD, 256), 256, 0, stream>>>(dw1, wb, GHD * GHD);
  gemm(xb, wb, hbuf, GHD, GHD);
  gnn_col_stats<<<GHD, 256, 0, stream>>>(hbuf, mean, rstd, N, GHD);
  gnn_bn_relu_cast<<<cdiv(N * GHD, 256), 256, 0, stream>>>(hbuf, mean, rstd, dg1, dbe1, xb, N * GHD, GHD);

  // ---- dense head 2 + final ----
  gnn_cast_f16<<<cdiv(GDIN * GHD, 256), 256, 0, stream>>>(dw2, wb, GDIN * GHD);
  gemm(xb, wb, hbuf, GHD, GDIN);
  gnn_final<<<cdiv(N * GDIN, 256), 256, 0, stream>>>(hbuf, db2, sf, (float*)d_out, N);
}
